// StackedGCN_22625887715494
// MI455X (gfx1250) — compile-verified, matrix operand order checked by reading, not run on hardware
//
#include <hip/hip_runtime.h>
#include <hip/hip_bf16.h>

// ---------------- problem constants (from reference) ----------------
#define N_NODES 100000
#define N_EDGES 1600000
// layer widths: 500 -> 128 -> 128 -> 64

typedef __attribute__((ext_vector_type(16))) __bf16 v16bf;
typedef __attribute__((ext_vector_type(8)))  __bf16 v8bf;
typedef __attribute__((ext_vector_type(4)))  __bf16 v4bf;
typedef __attribute__((ext_vector_type(8)))  float  v8f;

// =====================================================================
// Degree / norm precompute
// =====================================================================
__global__ __launch_bounds__(256)
void deg_init_kernel(float* __restrict__ deg) {
    int i = blockIdx.x * blockDim.x + threadIdx.x;
    if (i < N_NODES) deg[i] = 1.0f;            // self loop
}

__global__ __launch_bounds__(256)
void deg_count_kernel(const long long* __restrict__ dst, float* __restrict__ deg) {
    int e = blockIdx.x * blockDim.x + threadIdx.x;
    if (e < N_EDGES) atomicAdd(&deg[(int)dst[e]], 1.0f);
}

__global__ __launch_bounds__(256)
void dinv_kernel(float* __restrict__ deg) {
    int i = blockIdx.x * blockDim.x + threadIdx.x;
    if (i < N_NODES) deg[i] = rsqrtf(deg[i]);  // deg >= 1 always (self loop)
}

// =====================================================================
// One-time weight prep: W[K x F] f32 -> Wt[F x Kp] bf16 (transposed,
// K zero-padded to Kp = multiple of 32). Removes all conversion and
// tail guards from the GEMM B-staging path.
// =====================================================================
__global__ __launch_bounds__(256)
void cvt_w_kernel(const float* __restrict__ W, __bf16* __restrict__ Wt,
                  int K, int F, int Kp)
{
    int t = blockIdx.x * blockDim.x + threadIdx.x;
    if (t >= F * Kp) return;
    int f = t / Kp, k = t % Kp;
    float v = (k < K) ? W[(size_t)k * F + f] : 0.0f;
    Wt[(size_t)f * Kp + k] = (__bf16)v;
}

// =====================================================================
// bf16 WMMA GEMM:  C[M x F] = A[M x K] @ W[K x F]
// Block = 256 threads (8 waves). Tile BM=32 x BN, KT=32.
// A (f32) is staged via b128 loads + packed bf16 cvt into LDS.
// B (pre-converted bf16, transposed) is staged with
// GLOBAL_LOAD_ASYNC_TO_LDS_B128 (ASYNCcnt), no VGPR round trip.
// Wave (wm, wn) owns a 16 x (16*NACC) C strip; A fragment reused
// across NACC v_wmma_f32_16x16x32_bf16 per k-step.
// =====================================================================
template<int K, int F, int BN>
__global__ __launch_bounds__(256)
void gemm_wmma_kernel(const float* __restrict__ A, const __bf16* __restrict__ Wt,
                      float* __restrict__ C)
{
    constexpr int KT   = 32;
    constexpr int Kp   = (K + KT - 1) / KT * KT;  // padded K of Wt
    constexpr int NACC = BN / 64;                 // C tiles per wave (2 or 1)
    constexpr int NB   = BN / 64;                 // async b128 per thread

    __shared__ __bf16 As[32][KT];                 // [m][k]
    __shared__ __bf16 Bt[BN][KT];                 // [n][k]

    const int tid  = threadIdx.x;
    const int wave = tid >> 5;
    const int lane = tid & 31;
    const int hlf  = lane >> 4;                   // lane half selects K sub-block
    const int l16  = lane & 15;
    const int wm   = wave >> 2;                   // 0..1 (row tile)
    const int wn   = wave & 3;                    // 0..3 (col strip)

    const int rowBase = blockIdx.x * 32;          // M = 100000 = 3125*32, exact
    const int colBase = blockIdx.y * BN;

    // ---- A staging coordinates (one float4 per thread) ----
    const int ar  = tid >> 3;                     // A row 0..31
    const int ac4 = (tid & 7) << 2;               // A k offset, float4 granular

    // ---- B staging: per-thread 16B segments, fixed LDS dest + W source ----
    unsigned      btLds[NB];
    const __bf16* wtSrc[NB];
    #pragma unroll
    for (int i = 0; i < NB; ++i) {
        int u = tid * NB + i;                     // 16B unit index in tile
        int r = u >> 2;                           // Bt row (column n), 4 units/row
        int o = (u & 3) * 8;                      // bf16 offset within row
        btLds[i] = (unsigned)(uintptr_t)&Bt[r][o];        // LDS byte address
        wtSrc[i] = Wt + (size_t)(colBase + r) * Kp + o;
    }

    v8f acc[NACC];
    #pragma unroll
    for (int n = 0; n < NACC; ++n) acc[n] = (v8f){};

    auto stageA = [&](int k0, bool guard) {
        float4 v;
        if (!guard || (k0 + ac4 + 3 < K)) {       // K % 4 == 0: full or empty
            v = *(const float4*)(A + (size_t)(rowBase + ar) * K + k0 + ac4);
        } else {
            v = make_float4(0.f, 0.f, 0.f, 0.f);
        }
        *(v4bf*)(&As[ar][ac4]) =
            (v4bf){(__bf16)v.x, (__bf16)v.y, (__bf16)v.z, (__bf16)v.w};
    };

    auto stageB = [&](int k0) {                   // async DMA global -> LDS
        #pragma unroll
        for (int i = 0; i < NB; ++i) {
            asm volatile("global_load_async_to_lds_b128 %0, %1, off"
                         :: "v"(btLds[i]),
                            "v"((unsigned long long)(uintptr_t)(wtSrc[i] + k0))
                         : "memory");
        }
    };

    auto compute = [&]() {
        // A fragment: row = wm*16+l16, K = {hlf*8..+7} U {16+hlf*8..+7}
        const __bf16* ap = &As[wm * 16 + l16][hlf * 8];
        v8bf a0 = *(const v8bf*)(ap);
        v8bf a1 = *(const v8bf*)(ap + 16);
        v16bf a = __builtin_shufflevector(a0, a1,
                    0,1,2,3,4,5,6,7,8,9,10,11,12,13,14,15);
        #pragma unroll
        for (int n = 0; n < NACC; ++n) {
            const __bf16* bp = &Bt[(wn * NACC + n) * 16 + l16][hlf * 8];
            v8bf b0 = *(const v8bf*)(bp);
            v8bf b1 = *(const v8bf*)(bp + 16);
            v16bf b = __builtin_shufflevector(b0, b1,
                        0,1,2,3,4,5,6,7,8,9,10,11,12,13,14,15);
            acc[n] = __builtin_amdgcn_wmma_f32_16x16x32_bf16(
                         false, a, false, b, (short)0, acc[n], false, false);
        }
    };

    constexpr int KF = (K / KT) * KT;
    for (int k0 = 0; k0 < KF; k0 += KT) {
        stageB(k0);
        stageA(k0, false);
        if (k0 + KT < K)                          // prefetch next A tile
            __builtin_prefetch(&A[(size_t)(rowBase + ar) * K + k0 + KT + ac4], 0, 1);
        asm volatile("s_wait_asynccnt 0x0" ::: "memory");
        __syncthreads();
        compute();
        __syncthreads();
    }
    if (K % KT) {                                 // tail: A guarded, B zero-padded
        stageB(KF);
        stageA(KF, true);
        asm volatile("s_wait_asynccnt 0x0" ::: "memory");
        __syncthreads();
        compute();
    }

    // C layout: VGPR i -> row hlf*8+i, col = l16 within a 16x16 tile
    const int row0 = rowBase + wm * 16 + hlf * 8;
    #pragma unroll
    for (int n = 0; n < NACC; ++n) {
        const int col = colBase + (wn * NACC + n) * 16 + l16;
        #pragma unroll
        for (int i = 0; i < 8; ++i)
            C[(size_t)(row0 + i) * F + col] = acc[n][i];
    }
}

// =====================================================================
// Propagation: out = sum_{e: dst=i} norm_e * x[src_e]  (+ self loop)
// =====================================================================
// self-loop term initializes out (full overwrite, float4 vectorized)
__global__ __launch_bounds__(256)
void selfloop_init_kernel(const float* __restrict__ x, float* __restrict__ out,
                          const float* __restrict__ dinv, int fshift /*log2 F*/)
{
    size_t t = (size_t)blockIdx.x * blockDim.x + threadIdx.x;
    if (t >= ((size_t)N_NODES << fshift) >> 2) return;
    size_t t4 = t << 2;
    int i = (int)(t4 >> fshift);
    float d = dinv[i];
    float w = d * d;
    float4 v = *(const float4*)(x + t4);
    *(float4*)(out + t4) = make_float4(v.x * w, v.y * w, v.z * w, v.w * w);
}

// one thread per (edge, 4-float chunk): float4 gather + 4 f32 atomic scatter
__global__ __launch_bounds__(256)
void edge_scatter_kernel(const float* __restrict__ x, float* __restrict__ out,
                         const long long* __restrict__ src,
                         const long long* __restrict__ dst,
                         const float* __restrict__ dinv,
                         int fshift /*log2 F*/, int cshift /*log2 (F/4)*/)
{
    long long t = (long long)blockIdx.x * blockDim.x + threadIdx.x;
    long long e = t >> cshift;
    if (e >= N_EDGES) return;
    int c4 = ((int)t & ((1 << cshift) - 1)) << 2;
    int s = (int)src[e], d = (int)dst[e];
    float w = dinv[s] * dinv[d];
    const float4 v = *(const float4*)(x + ((size_t)s << fshift) + c4);
    float* o = out + ((size_t)d << fshift) + c4;
    atomicAdd(o + 0, v.x * w);
    atomicAdd(o + 1, v.y * w);
    atomicAdd(o + 2, v.z * w);
    atomicAdd(o + 3, v.w * w);
}

__global__ __launch_bounds__(256)
void bias_act_kernel(float* __restrict__ h, const float* __restrict__ b,
                     int fshift, int relu)
{
    size_t t = (size_t)blockIdx.x * blockDim.x + threadIdx.x;
    if (t >= ((size_t)N_NODES << fshift) >> 2) return;
    size_t t4 = t << 2;
    int f = (int)t4 & ((1 << fshift) - 1);
    float4 bv = *(const float4*)(b + f);
    float4 v  = *(const float4*)(h + t4);
    v.x += bv.x; v.y += bv.y; v.z += bv.z; v.w += bv.w;
    if (relu) {
        v.x = fmaxf(v.x, 0.f); v.y = fmaxf(v.y, 0.f);
        v.z = fmaxf(v.z, 0.f); v.w = fmaxf(v.w, 0.f);
    }
    *(float4*)(h + t4) = v;
}

// =====================================================================
// log_softmax over 64 classes: one wave32 per row, 2 elems per lane
// =====================================================================
__global__ __launch_bounds__(256)
void log_softmax64_kernel(float* __restrict__ out)
{
    int row  = blockIdx.x * 8 + (threadIdx.x >> 5);
    int lane = threadIdx.x & 31;
    if (row >= N_NODES) return;
    float* p = out + (size_t)row * 64;
    float x0 = p[lane], x1 = p[lane + 32];
    float m = fmaxf(x0, x1);
    #pragma unroll
    for (int off = 16; off >= 1; off >>= 1)
        m = fmaxf(m, __shfl_xor(m, off, 32));
    float s = __expf(x0 - m) + __expf(x1 - m);
    #pragma unroll
    for (int off = 16; off >= 1; off >>= 1)
        s += __shfl_xor(s, off, 32);
    float ls = __logf(s);
    p[lane]      = x0 - m - ls;
    p[lane + 32] = x1 - m - ls;
}

// =====================================================================
extern "C" void kernel_launch(void* const* d_in, const int* in_sizes, int n_in,
                              void* d_out, int out_size, void* d_ws, size_t ws_size,
                              hipStream_t stream)
{
    const float*     feat  = (const float*)d_in[0];       // [N, 500]
    const long long* edges = (const long long*)d_in[1];   // [2, E] int64
    const float*     W1    = (const float*)d_in[2];
    const float*     b1    = (const float*)d_in[3];
    const float*     W2    = (const float*)d_in[4];
    const float*     b2    = (const float*)d_in[5];
    const float*     W3    = (const float*)d_in[6];
    const float*     b3    = (const float*)d_in[7];
    float*           outp  = (float*)d_out;               // [N, 64]

    const long long* src = edges;
    const long long* dst = edges + N_EDGES;

    // workspace: dinv[N] | bufA[N*128] | bufB[N*128] | Wt1 | Wt2 | Wt3 (bf16)
    float*  dinv = (float*)d_ws;
    float*  bufA = dinv + N_NODES;
    float*  bufB = bufA + (size_t)N_NODES * 128;
    __bf16* Wt1  = (__bf16*)(bufB + (size_t)N_NODES * 128);  // 128 x 512
    __bf16* Wt2  = Wt1 + 128 * 512;                          // 128 x 128
    __bf16* Wt3  = Wt2 + 128 * 128;                          //  64 x 128

    const int T = 256;
    dim3 blk(T);

    // ---- normalization coefficients ----
    deg_init_kernel <<<(N_NODES + T - 1) / T, blk, 0, stream>>>(dinv);
    deg_count_kernel<<<(N_EDGES + T - 1) / T, blk, 0, stream>>>(dst, dinv);
    dinv_kernel     <<<(N_NODES + T - 1) / T, blk, 0, stream>>>(dinv);

    // ---- weight prep: f32 [K,F] -> bf16 transposed padded [F,Kp] ----
    cvt_w_kernel<<<(128 * 512 + T - 1) / T, blk, 0, stream>>>(W1, Wt1, 500, 128, 512);
    cvt_w_kernel<<<(128 * 128 + T - 1) / T, blk, 0, stream>>>(W2, Wt2, 128, 128, 128);
    cvt_w_kernel<<<( 64 * 128 + T - 1) / T, blk, 0, stream>>>(W3, Wt3, 128,  64, 128);

    const int mBlocks = N_NODES / 32;                 // 3125 (exact)

    // ---- layer 1: X@W1 -> propagate -> +b1, relu ----
    gemm_wmma_kernel<500, 128, 128><<<dim3(mBlocks, 1), blk, 0, stream>>>(feat, Wt1, bufA);
    {
        size_t n4 = (size_t)N_NODES * 128 / 4;
        selfloop_init_kernel<<<(unsigned)((n4 + T - 1) / T), blk, 0, stream>>>(bufA, bufB, dinv, 7);
        long long work = (long long)N_EDGES * 32;     // F/4 = 32 chunks
        edge_scatter_kernel<<<(unsigned)((work + T - 1) / T), blk, 0, stream>>>(
            bufA, bufB, src, dst, dinv, 7, 5);
        bias_act_kernel<<<(unsigned)((n4 + T - 1) / T), blk, 0, stream>>>(bufB, b1, 7, 1);
    }

    // ---- layer 2: h1@W2 -> propagate -> +b2, relu ----
    gemm_wmma_kernel<128, 128, 128><<<dim3(mBlocks, 1), blk, 0, stream>>>(bufB, Wt2, bufA);
    {
        size_t n4 = (size_t)N_NODES * 128 / 4;
        selfloop_init_kernel<<<(unsigned)((n4 + T - 1) / T), blk, 0, stream>>>(bufA, bufB, dinv, 7);
        long long work = (long long)N_EDGES * 32;
        edge_scatter_kernel<<<(unsigned)((work + T - 1) / T), blk, 0, stream>>>(
            bufA, bufB, src, dst, dinv, 7, 5);
        bias_act_kernel<<<(unsigned)((n4 + T - 1) / T), blk, 0, stream>>>(bufB, b2, 7, 1);
    }

    // ---- layer 3: h2@W3 -> propagate into d_out -> +b3 (no relu) ----
    gemm_wmma_kernel<128, 64, 64><<<dim3(mBlocks, 1), blk, 0, stream>>>(bufB, Wt3, bufA);
    {
        size_t n4 = (size_t)N_NODES * 64 / 4;
        selfloop_init_kernel<<<(unsigned)((n4 + T - 1) / T), blk, 0, stream>>>(bufA, outp, dinv, 6);
        long long work = (long long)N_EDGES * 16;     // F/4 = 16 chunks
        edge_scatter_kernel<<<(unsigned)((work + T - 1) / T), blk, 0, stream>>>(
            bufA, outp, src, dst, dinv, 6, 4);
        bias_act_kernel<<<(unsigned)((n4 + T - 1) / T), blk, 0, stream>>>(outp, b3, 6, 0);
    }

    // ---- log_softmax over 64 classes ----
    log_softmax64_kernel<<<N_NODES / 8, blk, 0, stream>>>(outp);
}
